// ApproxSymmetricNet_38019050504697
// MI455X (gfx1250) — matrix operand order repaired and checked by reading, not compiled, not run on hardware
//
#include <hip/hip_runtime.h>

// ---- problem constants (from reference) ----
#define BATCH   128
#define NSITES  16384
#define NPLAQ   16384
#define KCHI    9
#define PSZ     4
#define KOMG    5
#define ROWSZ   1024            // 8 comps (4 re + 4 im) * 128 batch
#define WILSON_RESCALE 31.6227766017f

#define SPB 8                   // sites per block in chi kernel
#define PPB 16                  // plaquettes per block in omega kernel

typedef __attribute__((ext_vector_type(16))) _Float16 v16h;
typedef __attribute__((ext_vector_type(8)))  float    v8f;
typedef __attribute__((ext_vector_type(4)))  int      v4i;

union Pack8 { _Float16 h[8]; uint4 u; };

#if __has_builtin(__builtin_amdgcn_global_load_async_to_lds_b128) && \
    __has_builtin(__builtin_amdgcn_s_wait_asynccnt)
#define HAVE_ASYNC_LDS 1
typedef __attribute__((address_space(1))) v4i* v4i_gptr;
typedef __attribute__((address_space(3))) v4i* v4i_lptr;
#endif

// complex tanh: tanh(a+ib) = (sinh 2a + i sin 2b) / (cosh 2a + cos 2b)
// (v_rcp_f32 instead of correctly-rounded divide: epilogue is not the
// precision-critical path and the IEEE expansion costs ~6 extra VALU each)
__device__ inline void ctanhf(float a, float b, float& tr, float& ti) {
    float e2a  = __expf(2.0f * a);
    float em2a = __expf(-2.0f * a);
    float sh = 0.5f * (e2a - em2a);
    float ch = 0.5f * (e2a + em2a);
    float s, c;
    __sincosf(2.0f * b, &s, &c);
    float inv = __builtin_amdgcn_rcpf(ch + c);
    tr = sh * inv;
    ti = s * inv;
}

// ---------------- A-operand packing (once per block; ISA 7.12.2) ------------
// A (16x32, MxK): lane l -> M = l%16 ; element i -> K = i + (l>=16?16:0) (contiguous K order used by this code; B matches)

// chi A: rows 0..3 = Re(chi_w[c][k]), rows 4..7 = Im(chi_w[c][k]); K = tap (<9)
__device__ inline v16h pack_chiA(const float* __restrict__ chi_w, int lane) {
    v16h a;
    int M = lane & 15;
    int kadd = (lane >= 16) ? 16 : 0;
#pragma unroll
    for (int i = 0; i < 16; ++i) {
        int K = i + kadd;
        float v = 0.0f;
        if (M < 8 && K < KCHI) {
            int c = M & 3;
            v = chi_w[(c * KCHI + K) * 2 + (M >= 4 ? 1 : 0)];
        }
        a[i] = (_Float16)v;
    }
    return a;
}

// omega A: global K' = kk*8 + comp (comp 0..3 Re ch, 4..7 Im ch), K' < 40.
// rows 0..3 (Re out o): comp<4 -> Re w[o][i][kk] ; comp>=4 -> -Im w
// rows 4..7 (Im out o): comp<4 -> Im w           ; comp>=4 ->  Re w
__device__ inline v16h pack_omgA(const float* __restrict__ omega_w, int lane, int koff) {
    v16h a;
    int M = lane & 15;
    int kadd = (lane >= 16) ? 16 : 0;
#pragma unroll
    for (int i = 0; i < 16; ++i) {
        int K = koff + i + kadd;
        float v = 0.0f;
        if (M < 8 && K < 8 * KOMG) {
            int kk = K >> 3, comp = K & 7;
            int o = M & 3, ci = comp & 3;
            const float* wp = omega_w + ((o * 4 + ci) * KOMG + kk) * 2;
            float wre = wp[0], wim = wp[1];
            if (M < 4) v = (comp < 4) ? wre : -wim;
            else       v = (comp < 4) ? wim :  wre;
        }
        a[i] = (_Float16)v;
    }
    return a;
}

// ---------------- kernels ---------------------------------------------------

// x (B, N) row-major -> xT (N, B): make batch the contiguous (coalesced) axis
__global__ void transpose_kernel(const float* __restrict__ x, float* __restrict__ xT) {
    __shared__ float tile[32][33];
    int tx = threadIdx.x, ty = threadIdx.y;
    int n0 = blockIdx.x * 32, b0 = blockIdx.y * 32;
#pragma unroll
    for (int j = 0; j < 32; j += 8)
        tile[ty + j][tx] = x[(size_t)(b0 + ty + j) * NSITES + n0 + tx];
    __syncthreads();
#pragma unroll
    for (int j = 0; j < 32; j += 8)
        xT[(size_t)(n0 + ty + j) * BATCH + b0 + tx] = tile[tx][ty + j];
}

// stage 1: per site, gather 9 x-rows, WMMA against chi weights, complex tanh,
// store h1[n][comp][b] (comp 0..3 = Re ch, 4..7 = Im ch).
// LDS layout: xg[b][32] f16, K-major, taps 9..31 zero-padded (branch-free
// B-operand = one aligned 32B contiguous LDS vector load per lane).
__global__ void __launch_bounds__(128)
chi_kernel(const float* __restrict__ xT, const int* __restrict__ chi_idx,
           const float* __restrict__ chi_w, const float* __restrict__ chi_b,
           float* __restrict__ h1) {
    __shared__ uint4 xgraw[BATCH * 4];                 // 128 rows * 64 B = 8 KB
    char* xg = (char*)xgraw;
    const int t = threadIdx.x;
    const int lane = t & 31;
    const int wave = t >> 5;
    const v16h A = pack_chiA(chi_w, lane);
    float cbr[4], cbi[4];
#pragma unroll
    for (int o = 0; o < 4; ++o) { cbr[o] = chi_b[2 * o]; cbi[o] = chi_b[2 * o + 1]; }

    // zero pad: taps 8..31 of this thread's row (tap 8 rewritten per site)
    uint4 z = {0, 0, 0, 0};
    xgraw[t * 4 + 1] = z;
    xgraw[t * 4 + 2] = z;
    xgraw[t * 4 + 3] = z;

    const int n0 = blockIdx.x * SPB;
    for (int si = 0; si < SPB; ++si) {
        const int n = n0 + si;
        __syncthreads();
        // gather this thread's batch column: 9 coalesced loads
        float v[KCHI];
#pragma unroll
        for (int k = 0; k < KCHI; ++k)
            v[k] = xT[(size_t)chi_idx[n * KCHI + k] * BATCH + t];
        Pack8 pk;
#pragma unroll
        for (int k = 0; k < 8; ++k) pk.h[k] = (_Float16)v[k];
        xgraw[t * 4] = pk.u;                                   // taps 0..7
        *(_Float16*)(xg + t * 64 + 16) = (_Float16)v[8];       // tap 8
        __syncthreads();
#pragma unroll
        for (int j = 0; j < 2; ++j) {
            const int bt = wave * 2 + j;
            const int b = bt * 16 + (lane & 15);
            // B operand: 16 contiguous f16 (K = 0..15 or 16..31), aligned 32B
            v16h Bv = *(const v16h*)(xg + b * 64 + ((lane >= 16) ? 32 : 0));
            v8f c = {};
            c = __builtin_amdgcn_wmma_f32_16x16x32_f16(false, A, false, Bv,
                                                       (short)0, c, false, false);
            if (lane < 16) {
                const int bb = bt * 16 + lane;
                float* outp = h1 + (size_t)n * ROWSZ;
#pragma unroll
                for (int o = 0; o < 4; ++o) {
                    float tr, ti;
                    ctanhf(c[o] + cbr[o], c[o + 4] + cbi[o], tr, ti);
                    outp[o * BATCH + bb]       = tr;
                    outp[(o + 4) * BATCH + bb] = ti;
                }
            }
        }
    }
}

// stage 2: wilson map. per (p,b): comp products over 4 gathered sites,
// store h2[p][b][comp] (comp contiguous -> one uint4 store per thread).
__global__ void __launch_bounds__(128)
wilson_kernel(const float* __restrict__ h1, const int* __restrict__ pidx,
              _Float16* __restrict__ h2) {
    const int p = blockIdx.x, t = threadIdx.x;
    const float* r0 = h1 + (size_t)pidx[p * PSZ + 0] * ROWSZ;
    const float* r1 = h1 + (size_t)pidx[p * PSZ + 1] * ROWSZ;
    const float* r2 = h1 + (size_t)pidx[p * PSZ + 2] * ROWSZ;
    const float* r3 = h1 + (size_t)pidx[p * PSZ + 3] * ROWSZ;
    Pack8 pk;
#pragma unroll
    for (int comp = 0; comp < 8; ++comp) {
        const int off = comp * BATCH + t;
        float v = WILSON_RESCALE * r0[off] * r1[off] * r2[off] * r3[off];
        pk.h[comp] = (_Float16)v;
    }
    ((uint4*)(h2 + (size_t)p * ROWSZ))[t] = pk.u;
}

// stage 3: per plaquette, stage 5 gathered h2 rows into LDS as g2[b][64] f16
// (K' = kk*8+comp contiguous, K'=40..63 zero-padded) via async LDS-DMA
// (per-lane global addr -> per-lane LDS addr, ASYNCcnt-tracked), then
// 2x WMMA against constant packed omega weights, complex tanh, reduce over
// (p,o), atomic add into out.
__global__ void __launch_bounds__(128)
omega_kernel(const _Float16* __restrict__ h2, const int* __restrict__ omg_idx,
             const float* __restrict__ omega_w, const float* __restrict__ omega_b,
             float* __restrict__ out) {
    __shared__ uint4 g2raw[BATCH * 8];                 // 128 rows * 128 B = 16 KB
    char* g2 = (char*)g2raw;
    const int t = threadIdx.x;
    const int lane = t & 31;
    const int wave = t >> 5;
    const v16h A0 = pack_omgA(omega_w, lane, 0);
    const v16h A1 = pack_omgA(omega_w, lane, 32);
    float obr[4], obi[4];
#pragma unroll
    for (int o = 0; o < 4; ++o) { obr[o] = omega_b[2 * o]; obi[o] = omega_b[2 * o + 1]; }

    // zero pad: K' = 40..63 of this thread's row (stays zero across plaquettes)
    uint4 z = {0, 0, 0, 0};
    g2raw[t * 8 + 5] = z;
    g2raw[t * 8 + 6] = z;
    g2raw[t * 8 + 7] = z;

    float accR[2] = {0.0f, 0.0f}, accI[2] = {0.0f, 0.0f};
    const int p0 = blockIdx.x * PPB;
    for (int pi = 0; pi < PPB; ++pi) {
        const int p = p0 + pi;
        __syncthreads();
#pragma unroll
        for (int kk = 0; kk < KOMG; ++kk) {
            const int s = omg_idx[p * KOMG + kk];
            // one 16B row slice (8 comps of this thread's b) -> LDS
            const uint4* src = (const uint4*)(h2 + (size_t)s * ROWSZ) + t;
            uint4* dst = &g2raw[t * 8 + kk];
#ifdef HAVE_ASYNC_LDS
            __builtin_amdgcn_global_load_async_to_lds_b128((v4i_gptr)src, (v4i_lptr)dst, 0, 0);
#else
            *dst = *src;
#endif
        }
#ifdef HAVE_ASYNC_LDS
        __builtin_amdgcn_s_wait_asynccnt(0);
#endif
        __syncthreads();
#pragma unroll
        for (int j = 0; j < 2; ++j) {
            const int bt = wave * 2 + j;
            const int b = bt * 16 + (lane & 15);
            const int hi = (lane >= 16) ? 32 : 0;
            v16h B0 = *(const v16h*)(g2 + b * 128 + hi);        // K' 0..31
            v16h B1 = *(const v16h*)(g2 + b * 128 + 64 + hi);   // K' 32..63
            v8f c = {};
            c = __builtin_amdgcn_wmma_f32_16x16x32_f16(false, A0, false, B0,
                                                       (short)0, c, false, false);
            c = __builtin_amdgcn_wmma_f32_16x16x32_f16(false, A1, false, B1,
                                                       (short)0, c, false, false);
            if (lane < 16) {
#pragma unroll
                for (int o = 0; o < 4; ++o) {
                    float tr, ti;
                    ctanhf(c[o] + obr[o], c[o + 4] + obi[o], tr, ti);
                    accR[j] += tr;
                    accI[j] += ti;
                }
            }
        }
    }
    if (lane < 16) {
#pragma unroll
        for (int j = 0; j < 2; ++j) {
            const int b = (wave * 2 + j) * 16 + lane;
            atomicAdd(&out[2 * b],     accR[j]);
            atomicAdd(&out[2 * b + 1], accI[j]);
        }
    }
}

// ---------------- launch ----------------------------------------------------

extern "C" void kernel_launch(void* const* d_in, const int* in_sizes, int n_in,
                              void* d_out, int out_size, void* d_ws, size_t ws_size,
                              hipStream_t stream) {
    const float* x       = (const float*)d_in[0];
    const int*   chi_idx = (const int*)d_in[1];
    const int*   pidx    = (const int*)d_in[3];
    const int*   oidx    = (const int*)d_in[5];
    const float* chi_w   = (const float*)d_in[7];
    const float* chi_b   = (const float*)d_in[8];
    const float* omega_w = (const float*)d_in[9];
    const float* omega_b = (const float*)d_in[10];
    float* out = (float*)d_out;

    char* ws = (char*)d_ws;
    float*    xT = (float*)ws;                               // 8 MB
    float*    h1 = (float*)(ws + ((size_t)8  << 20));        // 64 MB
    _Float16* h2 = (_Float16*)(ws + ((size_t)72 << 20));     // 32 MB

    (void)hipMemsetAsync(d_out, 0, (size_t)out_size * sizeof(float), stream);

    transpose_kernel<<<dim3(NSITES / 32, BATCH / 32), dim3(32, 8), 0, stream>>>(x, xT);
    chi_kernel<<<NSITES / SPB, 128, 0, stream>>>(xT, chi_idx, chi_w, chi_b, h1);
    wilson_kernel<<<NPLAQ, 128, 0, stream>>>(h1, pidx, h2);
    omega_kernel<<<NPLAQ / PPB, 128, 0, stream>>>(h2, oidx, omega_w, omega_b, out);
}
